// GD3PM_37022618092063
// MI455X (gfx1250) — matrix operand order repaired and checked by reading, not compiled, not run on hardware
//
#include <hip/hip_runtime.h>
#include <hip/hip_bf16.h>

// ---------------- model constants ----------------
constexpr int B_   = 8;
constexpr int N_   = 64;
constexpr int NH_  = 576;   // node hidden
constexpr int EH_  = 128;   // edge hidden
constexpr int TH_  = 128;   // time hidden
constexpr int H_   = 12;    // heads
constexpr int D_   = 48;    // per-head dim
constexpr int L_   = 48;    // layers
constexpr int X_   = NH_ + TH_;          // 704 concat dim
constexpr int RN_  = B_ * N_;            // 512 node rows
constexpr int RE_  = B_ * N_ * N_;       // 32768 edge rows
constexpr int NODE_DIM_ = 45;
constexpr int EDGE_DIM_ = 5;

typedef __attribute__((ext_vector_type(16))) __bf16 v16bf;
typedef __attribute__((ext_vector_type(8)))  float  v8f;

__device__ __forceinline__ float leaky_f(float x) { return x >= 0.f ? x : 0.05f * x; }

// ---------------------------------------------------------------------------
// Weight prep: f32 W[K x N] (row-major, batched) -> zero-padded bf16
// W^T [Npad x Kp] so GEMM B-fragments are contiguous 32B per lane.
// ---------------------------------------------------------------------------
__global__ void wprep(const float* __restrict__ W, __bf16* __restrict__ Wt,
                      int K, int N, int Kp, int Np, int batch) {
  long t = (long)blockIdx.x * blockDim.x + threadIdx.x;
  long per = (long)Np * Kp;
  if (t >= per * batch) return;
  int l = (int)(t / per);
  long r = t % per;
  int n = (int)(r / Kp);
  int k = (int)(r % Kp);
  float v = (k < K && n < N) ? W[(size_t)l * K * N + (size_t)k * N + n] : 0.f;
  Wt[t] = (__bf16)v;
}

// ------------- pad activations [rows x K] -> [rows x Kp] (zero fill) --------
__global__ void apad(const float* __restrict__ src, float* __restrict__ dst,
                     int K, int Kp, long rows) {
  long t = (long)blockIdx.x * blockDim.x + threadIdx.x;
  if (t >= rows * Kp) return;
  long r = t / Kp;
  int  k = (int)(t % Kp);
  dst[t] = (k < K) ? src[(size_t)r * K + k] : 0.f;
}

// ---------------------------------------------------------------------------
// GEMM: C[M x N] = act(A[M x Kp] @ W + bias) with W pre-transposed bf16
// [Npad x Kp].  One wave = 32 rows x 64 cols (2 M-tiles x 4 N-tiles), bf16
// WMMA f32-accumulate, branch-free K-loop; 8 WMMAs per K-step reuse the 4 B
// fragments across both M-tiles and both A fragments across all 4 N-tiles.
//   A frag (16-bit A 16x32):  lane=half*16+m ; elem i -> k=i+16*half
//   B frag (32x16):           lane=half*16+n ; elem i -> k=i+16*half
//   C/D f32 16x16: lane col n=lane&15 ; elem r -> row r+8*(lane>>4)
// Kp % 32 == 0, Npad % 64 == 0, M % 32 == 0; only the store guards col < N.
// ---------------------------------------------------------------------------
__global__ void __launch_bounds__(256)
gemm_bf16(const float* __restrict__ A, const __bf16* __restrict__ Wt,
          const float* __restrict__ bias, float* __restrict__ C,
          int M, int Kp, int Np, int N, int leaky_out) {
  const int lane = threadIdx.x & 31;
  const int wave = threadIdx.x >> 5;
  const int half = lane >> 4;
  const int l15  = lane & 15;

  const int groupsN = Np >> 6;
  const int pairsM  = M >> 5;
  const int tile = blockIdx.x * (blockDim.x >> 5) + wave;
  if (tile >= pairsM * groupsN) return;   // wave-uniform exit, EXEC stays full

  const int tm = tile / groupsN;
  const int gn = tile % groupsN;
  const int rowBase = tm << 5;
  const int colBase = gn << 6;

  const float*  arow0 = A + (size_t)(rowBase + l15) * Kp + 16 * half;
  const float*  arow1 = arow0 + (size_t)16 * Kp;
  const __bf16* bbase = Wt + (size_t)(colBase + l15) * Kp + 16 * half;
  const size_t  bstep = (size_t)16 * Kp;   // per N-tile

  v8f acc[8] = {};

  for (int k0 = 0; k0 < Kp; k0 += 32) {
    if (k0 + 32 < Kp) {
      __builtin_prefetch(arow0 + k0 + 32, 0, 3);
      __builtin_prefetch(arow1 + k0 + 32, 0, 3);
      __builtin_prefetch(bbase + k0 + 32, 0, 3);
    }
    // ---- all loads for this K-step up front (12x b128-class) ----
    const float4* p0  = (const float4*)(arow0 + k0);
    const float4* p0b = (const float4*)(arow0 + k0 + 8);
    const float4* p1  = (const float4*)(arow1 + k0);
    const float4* p1b = (const float4*)(arow1 + k0 + 8);
    float4 x0 = p0[0],  x1 = p0[1],  x2 = p0b[0], x3 = p0b[1];
    float4 y0 = p1[0],  y1 = p1[1],  y2 = p1b[0], y3 = p1b[1];
    v16bf bv[4];
#pragma unroll
    for (int t = 0; t < 4; ++t)
      bv[t] = *(const v16bf*)(bbase + (size_t)t * bstep + k0);

    // ---- native packed converts f32 -> bf16 ----
    float af0[16] = {x0.x, x0.y, x0.z, x0.w, x1.x, x1.y, x1.z, x1.w,
                     x2.x, x2.y, x2.z, x2.w, x3.x, x3.y, x3.z, x3.w};
    float af1[16] = {y0.x, y0.y, y0.z, y0.w, y1.x, y1.y, y1.z, y1.w,
                     y2.x, y2.y, y2.z, y2.w, y3.x, y3.y, y3.z, y3.w};
    v16bf av0, av1;
#pragma unroll
    for (int i = 0; i < 16; ++i) { av0[i] = (__bf16)af0[i]; av1[i] = (__bf16)af1[i]; }

    // ---- 8 WMMAs ----
#pragma unroll
    for (int t = 0; t < 4; ++t) {
      acc[2 * t] = __builtin_amdgcn_wmma_f32_16x16x32_bf16(
          false, av0, false, bv[t], (short)0, acc[2 * t], false, false);
      acc[2 * t + 1] = __builtin_amdgcn_wmma_f32_16x16x32_bf16(
          false, av1, false, bv[t], (short)0, acc[2 * t + 1], false, false);
    }
  }

  // ---- epilogue: bias + optional leaky; only ragged-N tiles guard ----
#pragma unroll
  for (int t = 0; t < 4; ++t) {
    int col = colBase + (t << 4) + l15;
    if (col < N) {
      float bval = bias[col];
#pragma unroll
      for (int r = 0; r < 8; ++r) {
        int row0 = rowBase + r + 8 * half;
        float v0 = acc[2 * t][r] + bval;
        float v1 = acc[2 * t + 1][r] + bval;
        if (leaky_out) { v0 = leaky_f(v0); v1 = leaky_f(v1); }
        C[(size_t)row0 * N + col] = v0;
        C[(size_t)(row0 + 16) * N + col] = v1;
      }
    }
  }
}

// ---------------- time embedding: one-hot(t) @ W + b == row gather ----------
__global__ void time_embed(const int* __restrict__ ts, const float* __restrict__ w,
                           const float* __restrict__ b, float* __restrict__ out) {
  int t = blockIdx.x * blockDim.x + threadIdx.x;
  if (t >= B_ * TH_) return;
  int bb = t / TH_, c = t % TH_;
  out[t] = w[(size_t)ts[bb] * TH_ + c] + b[c];
}

// ---------------- x = concat(nodes, broadcast(times)) -----------------------
__global__ void concat_x(const float* __restrict__ n, const float* __restrict__ times,
                         float* __restrict__ x) {
  int t = blockIdx.x * blockDim.x + threadIdx.x;
  if (t >= RN_ * X_) return;
  int c = t % X_, row = t / X_, b = row / N_;
  x[t] = (c < NH_) ? n[(size_t)row * NH_ + c] : times[b * TH_ + (c - NH_)];
}

// ---- new_edges = att*em + att + ea ; logits = sum_d ; ne = leaky(new_edges)
__global__ void edge_fuse(const float* __restrict__ q, const float* __restrict__ k,
                          const float* __restrict__ em, const float* __restrict__ ea,
                          float* __restrict__ ne, float* __restrict__ logits) {
  int t = blockIdx.x * blockDim.x + threadIdx.x;
  if (t >= B_ * N_ * N_ * H_) return;
  int h = t % H_;
  int j = (t / H_) % N_;
  int i = (t / (H_ * N_)) % N_;
  int b = t / (H_ * N_ * N_);
  const float inv = 0.0416666667f;  // 1/sqrt(NH)=1/24
  const float* qp = q + (size_t)(b * N_ + i) * NH_ + h * D_;
  const float* kp = k + (size_t)(b * N_ + j) * NH_ + h * D_;
  size_t eoff = (size_t)((b * N_ + i) * N_ + j) * NH_ + h * D_;
  float s = 0.f;
  for (int d = 0; d < D_; ++d) {
    float att = qp[d] * kp[d] * inv;
    float v = att * em[eoff + d] + att + ea[eoff + d];
    s += v;
    ne[eoff + d] = leaky_f(v);
  }
  logits[t] = s;
}

// ---------------- softmax over j (in place), thread per (b,i,h) -------------
__global__ void softmax_j(float* __restrict__ logits) {
  int t = blockIdx.x * blockDim.x + threadIdx.x;
  if (t >= B_ * N_ * H_) return;
  int h = t % H_;
  int i = (t / H_) % N_;
  int b = t / (H_ * N_);
  float* p = logits + (size_t)(b * N_ + i) * N_ * H_ + h;
  float mx = -1e30f;
  for (int j = 0; j < N_; ++j) mx = fmaxf(mx, p[j * H_]);
  float s = 0.f;
  for (int j = 0; j < N_; ++j) { float e = __expf(p[j * H_] - mx); p[j * H_] = e; s += e; }
  float r = 1.f / s;
  for (int j = 0; j < N_; ++j) p[j * H_] *= r;
}

// ---------------- wv[b,i,h*48+d] = sum_j a[b,i,j,h] * v[b,j,h*48+d] ---------
__global__ void wv_apply(const float* __restrict__ a, const float* __restrict__ v,
                         float* __restrict__ out) {
  int t = blockIdx.x * blockDim.x + threadIdx.x;
  if (t >= RN_ * NH_) return;
  int c = t % NH_, row = t / NH_;
  int h = c / D_;
  int b = row / N_;
  const float* ap = a + (size_t)row * N_ * H_ + h;
  const float* vp = v + (size_t)(b * N_) * NH_ + c;
  float s = 0.f;
  for (int j = 0; j < N_; ++j) s += ap[j * H_] * vp[(size_t)j * NH_];
  out[t] = s;
}

// ------- out = maybe_leaky(LN(x + res) * g + b), one wave per row -----------
// Rows are staged x->LDS / res->LDS once via GLOBAL_LOAD_ASYNC_TO_LDS_B128
// (ASYNCcnt-tracked), then both LN passes read LDS instead of re-reading HBM.
// Each wave touches only its own LDS slice -> no barrier, s_wait_asynccnt 0.
__global__ void __launch_bounds__(256)
add_ln(const float* __restrict__ x, const float* __restrict__ res,
       const float* __restrict__ g, const float* __restrict__ bb,
       float* __restrict__ out, int rows, int width, int do_leaky) {
  __shared__ __align__(16) float smem[8][2][NH_];
  int wv = threadIdx.x >> 5;
  int gwave = (blockIdx.x * blockDim.x + threadIdx.x) >> 5;
  int lane = threadIdx.x & 31;
  if (gwave >= rows) return;
  const float* xp = x + (size_t)gwave * width;
  const float* rp = res + (size_t)gwave * width;
  float* lx = &smem[wv][0][0];
  float* lr = &smem[wv][1][0];

  // async stage: 16B per lane per issue (wave covers 512B per issue)
  for (int c = lane * 4; c < width; c += 128) {
    unsigned lxo = (unsigned)(size_t)(lx + c);   // low 32 bits == LDS offset
    unsigned lro = (unsigned)(size_t)(lr + c);
    unsigned long long gx = (unsigned long long)(const void*)(xp + c);
    unsigned long long gr = (unsigned long long)(const void*)(rp + c);
    asm volatile("global_load_async_to_lds_b128 %0, %1, off"
                 :: "v"(lxo), "v"(gx) : "memory");
    asm volatile("global_load_async_to_lds_b128 %0, %1, off"
                 :: "v"(lro), "v"(gr) : "memory");
  }
  asm volatile("s_wait_asynccnt 0x0" ::: "memory");

  float s = 0.f, s2 = 0.f;
  for (int c = lane; c < width; c += 32) {
    float v = lx[c] + lr[c];
    s += v; s2 += v * v;
  }
  for (int o = 16; o > 0; o >>= 1) {
    s  += __shfl_xor(s,  o, 32);
    s2 += __shfl_xor(s2, o, 32);
  }
  float m = s / width;
  float var = s2 / width - m * m;
  float inv = rsqrtf(var + 1e-5f);
  float* op = out + (size_t)gwave * width;
  for (int c = lane; c < width; c += 32) {
    float v = (lx[c] + lr[c] - m) * inv * g[c] + bb[c];
    if (do_leaky) v = leaky_f(v);
    op[c] = v;
  }
}

// ---------------------------------------------------------------------------
extern "C" void kernel_launch(void* const* d_in, const int* in_sizes, int n_in,
                              void* d_out, int out_size, void* d_ws, size_t ws_size,
                              hipStream_t stream) {
  (void)in_sizes; (void)n_in; (void)out_size; (void)ws_size;

  // setup_inputs() dict-insertion-order flattening; each _lin -> (w, b)
  const float* nodes_in = (const float*)d_in[0];
  const float* edges_in = (const float*)d_in[1];
  const int*   tstep    = (const int*)  d_in[2];
  const float* time_w  = (const float*)d_in[3];
  const float* time_b  = (const float*)d_in[4];
  const float* in_n1_w = (const float*)d_in[5];
  const float* in_n1_b = (const float*)d_in[6];
  const float* in_n2_w = (const float*)d_in[7];
  const float* in_n2_b = (const float*)d_in[8];
  const float* in_e1_w = (const float*)d_in[9];
  const float* in_e1_b = (const float*)d_in[10];
  const float* in_e2_w = (const float*)d_in[11];
  const float* in_e2_b = (const float*)d_in[12];
  const float* Lq_w   = (const float*)d_in[13];
  const float* Lq_b   = (const float*)d_in[14];
  const float* Lk_w   = (const float*)d_in[15];
  const float* Lk_b   = (const float*)d_in[16];
  const float* Lv_w   = (const float*)d_in[17];
  const float* Lv_b   = (const float*)d_in[18];
  const float* Lmul_w = (const float*)d_in[19];
  const float* Lmul_b = (const float*)d_in[20];
  const float* Ladd_w = (const float*)d_in[21];
  const float* Ladd_b = (const float*)d_in[22];
  const float* Lno_w  = (const float*)d_in[23];
  const float* Lno_b  = (const float*)d_in[24];
  const float* Leo_w  = (const float*)d_in[25];
  const float* Leo_b  = (const float*)d_in[26];
  const float* Lln1n_g = (const float*)d_in[27];
  const float* Lln1n_b = (const float*)d_in[28];
  const float* Lln1e_g = (const float*)d_in[29];
  const float* Lln1e_b = (const float*)d_in[30];
  const float* Lmn1_w = (const float*)d_in[31];
  const float* Lmn1_b = (const float*)d_in[32];
  const float* Lmn2_w = (const float*)d_in[33];
  const float* Lmn2_b = (const float*)d_in[34];
  const float* Lme1_w = (const float*)d_in[35];
  const float* Lme1_b = (const float*)d_in[36];
  const float* Lme2_w = (const float*)d_in[37];
  const float* Lme2_b = (const float*)d_in[38];
  const float* Lln2n_g = (const float*)d_in[39];
  const float* Lln2n_b = (const float*)d_in[40];
  const float* Lln2e_g = (const float*)d_in[41];
  const float* Lln2e_b = (const float*)d_in[42];
  const float* out_n1_w = (const float*)d_in[43];
  const float* out_n1_b = (const float*)d_in[44];
  const float* out_n2_w = (const float*)d_in[45];
  const float* out_n2_b = (const float*)d_in[46];
  const float* out_e1_w = (const float*)d_in[47];
  const float* out_e1_b = (const float*)d_in[48];
  const float* out_e2_w = (const float*)d_in[49];
  const float* out_e2_b = (const float*)d_in[50];

  // ---- workspace carve: f32 region, then bf16 transposed-weight region ----
  float* ws = (float*)d_ws;
  size_t off = 0;
  auto carve = [&](size_t n) { float* p = ws + off; off += n; return p; };
  float* times = carve((size_t)B_ * TH_);
  float* nbuf  = carve((size_t)RN_ * NH_);   // node state / an
  float* xbuf  = carve((size_t)RN_ * X_);    // concat x; also attn-out, mn
  float* qb    = carve((size_t)RN_ * NH_);
  float* kb    = carve((size_t)RN_ * NH_);
  float* vb    = carve((size_t)RN_ * NH_);
  float* wvb   = carve((size_t)RN_ * NH_);   // wv; also node-MLP hidden
  float* ebuf  = carve((size_t)RE_ * EH_);   // edge state; also me
  float* aebuf = carve((size_t)RE_ * EH_);   // eo out -> ae
  float* tebuf = carve((size_t)RE_ * EH_);   // edge-MLP hidden
  float* embuf = carve((size_t)RE_ * NH_);   // em, then leaky(new_edges) in place
  float* eabuf = carve((size_t)RE_ * NH_);   // ea
  float* logit = carve((size_t)B_ * N_ * N_ * H_);  // logits -> softmax in place
  float* npad  = carve((size_t)RN_ * 64);    // nodes_in padded K 45->64
  float* epad  = carve((size_t)RE_ * 64);    // edges_in padded K 5->64

  __bf16* wsb = (__bf16*)(ws + off);
  size_t boff = 0;
  auto carveb = [&](size_t n) { __bf16* p = wsb + boff; boff += n; return p; };
  __bf16* t_in1 = carveb((size_t)576 * 64);
  __bf16* t_in2 = carveb((size_t)576 * 576);
  __bf16* t_ie1 = carveb((size_t)128 * 64);
  __bf16* t_ie2 = carveb((size_t)128 * 128);
  __bf16* t_q   = carveb((size_t)L_ * 576 * 704);
  __bf16* t_k   = carveb((size_t)L_ * 576 * 704);
  __bf16* t_v   = carveb((size_t)L_ * 576 * 704);
  __bf16* t_mul = carveb((size_t)L_ * 576 * 128);
  __bf16* t_add = carveb((size_t)L_ * 576 * 128);
  __bf16* t_no  = carveb((size_t)L_ * 576 * 576);
  __bf16* t_eo  = carveb((size_t)L_ * 128 * 576);
  __bf16* t_m1  = carveb((size_t)L_ * 576 * 576);
  __bf16* t_m2  = carveb((size_t)L_ * 576 * 576);
  __bf16* t_e1  = carveb((size_t)L_ * 128 * 128);
  __bf16* t_e2  = carveb((size_t)L_ * 128 * 128);
  __bf16* t_on1 = carveb((size_t)576 * 576);
  __bf16* t_on2 = carveb((size_t)64 * 576);
  __bf16* t_oe1 = carveb((size_t)128 * 128);
  __bf16* t_oe2 = carveb((size_t)64 * 128);

  auto prep = [&](const float* W, __bf16* Wt, int K, int N, int Kp, int Np, int batch) {
    long total = (long)Np * Kp * batch;
    wprep<<<(int)((total + 255) / 256), 256, 0, stream>>>(W, Wt, K, N, Kp, Np, batch);
  };
  prep(in_n1_w, t_in1, 45, 576, 64, 576, 1);
  prep(in_n2_w, t_in2, 576, 576, 576, 576, 1);
  prep(in_e1_w, t_ie1, 5, 128, 64, 128, 1);
  prep(in_e2_w, t_ie2, 128, 128, 128, 128, 1);
  prep(Lq_w,   t_q,   704, 576, 704, 576, L_);
  prep(Lk_w,   t_k,   704, 576, 704, 576, L_);
  prep(Lv_w,   t_v,   704, 576, 704, 576, L_);
  prep(Lmul_w, t_mul, 128, 576, 128, 576, L_);
  prep(Ladd_w, t_add, 128, 576, 128, 576, L_);
  prep(Lno_w,  t_no,  576, 576, 576, 576, L_);
  prep(Leo_w,  t_eo,  576, 128, 576, 128, L_);
  prep(Lmn1_w, t_m1,  576, 576, 576, 576, L_);
  prep(Lmn2_w, t_m2,  576, 576, 576, 576, L_);
  prep(Lme1_w, t_e1,  128, 128, 128, 128, L_);
  prep(Lme2_w, t_e2,  128, 128, 128, 128, L_);
  prep(out_n1_w, t_on1, 576, 576, 576, 576, 1);
  prep(out_n2_w, t_on2, 576, 45, 576, 64, 1);
  prep(out_e1_w, t_oe1, 128, 128, 128, 128, 1);
  prep(out_e2_w, t_oe2, 128, 5, 128, 64, 1);
  apad<<<(int)(((long)RN_ * 64 + 255) / 256), 256, 0, stream>>>(nodes_in, npad, 45, 64, RN_);
  apad<<<(int)(((long)RE_ * 64 + 255) / 256), 256, 0, stream>>>(edges_in, epad, 5, 64, RE_);

  auto gemm = [&](const float* A, const __bf16* Wt, const float* bias, float* C,
                  int M, int Kp, int Np, int Nc, int lk) {
    int tiles = (M >> 5) * (Np >> 6);
    int blocks = (tiles + 7) >> 3;               // 8 waves per 256-thread block
    gemm_bf16<<<blocks, 256, 0, stream>>>(A, Wt, bias, C, M, Kp, Np, Nc, lk);
  };

  // ---- time embedding + input MLPs ----
  time_embed<<<(B_ * TH_ + 255) / 256, 256, 0, stream>>>(tstep, time_w, time_b, times);
  gemm(npad, t_in1, in_n1_b, xbuf, RN_, 64, 576, NH_, 1);
  gemm(xbuf, t_in2, in_n2_b, nbuf, RN_, 576, 576, NH_, 1);
  gemm(epad, t_ie1, in_e1_b, tebuf, RE_, 64, 128, EH_, 1);
  gemm(tebuf, t_ie2, in_e2_b, ebuf, RE_, 128, 128, EH_, 1);

  // ---- 48 transformer layers ----
  for (int l = 0; l < L_; ++l) {
    const __bf16* qw  = t_q   + (size_t)l * 576 * 704;
    const __bf16* kw  = t_k   + (size_t)l * 576 * 704;
    const __bf16* vw  = t_v   + (size_t)l * 576 * 704;
    const __bf16* mw  = t_mul + (size_t)l * 576 * 128;
    const __bf16* aw  = t_add + (size_t)l * 576 * 128;
    const __bf16* nw  = t_no  + (size_t)l * 576 * 576;
    const __bf16* ew  = t_eo  + (size_t)l * 128 * 576;
    const __bf16* m1w = t_m1  + (size_t)l * 576 * 576;
    const __bf16* m2w = t_m2  + (size_t)l * 576 * 576;
    const __bf16* e1w = t_e1  + (size_t)l * 128 * 128;
    const __bf16* e2w = t_e2  + (size_t)l * 128 * 128;
    const float* qbi = Lq_b   + (size_t)l * NH_;
    const float* kbi = Lk_b   + (size_t)l * NH_;
    const float* vbi = Lv_b   + (size_t)l * NH_;
    const float* mbi = Lmul_b + (size_t)l * NH_;
    const float* abi = Ladd_b + (size_t)l * NH_;
    const float* nbi = Lno_b  + (size_t)l * NH_;
    const float* ebi = Leo_b  + (size_t)l * EH_;
    const float* m1b = Lmn1_b + (size_t)l * NH_;
    const float* m2b = Lmn2_b + (size_t)l * NH_;
    const float* e1b = Lme1_b + (size_t)l * EH_;
    const float* e2b = Lme2_b + (size_t)l * EH_;
    const float* l1ng = Lln1n_g + (size_t)l * NH_;
    const float* l1nb = Lln1n_b + (size_t)l * NH_;
    const float* l1eg = Lln1e_g + (size_t)l * EH_;
    const float* l1eb = Lln1e_b + (size_t)l * EH_;
    const float* l2ng = Lln2n_g + (size_t)l * NH_;
    const float* l2nb = Lln2n_b + (size_t)l * NH_;
    const float* l2eg = Lln2e_g + (size_t)l * EH_;
    const float* l2eb = Lln2e_b + (size_t)l * EH_;

    // MHA
    concat_x<<<(RN_ * X_ + 255) / 256, 256, 0, stream>>>(nbuf, times, xbuf);
    gemm(xbuf, qw, qbi, qb, RN_, 704, 576, NH_, 0);
    gemm(xbuf, kw, kbi, kb, RN_, 704, 576, NH_, 0);
    gemm(xbuf, vw, vbi, vb, RN_, 704, 576, NH_, 0);
    gemm(ebuf, mw, mbi, embuf, RE_, 128, 576, NH_, 0);   // em
    gemm(ebuf, aw, abi, eabuf, RE_, 128, 576, NH_, 0);   // ea
    edge_fuse<<<(B_ * N_ * N_ * H_ + 255) / 256, 256, 0, stream>>>(
        qb, kb, embuf, eabuf, embuf /*ne in place*/, logit);
    softmax_j<<<(B_ * N_ * H_ + 255) / 256, 256, 0, stream>>>(logit);
    wv_apply<<<(RN_ * NH_ + 255) / 256, 256, 0, stream>>>(logit, vb, wvb);
    gemm(wvb,   nw, nbi, xbuf,  RN_, 576, 576, NH_, 0);  // node attn out
    gemm(embuf, ew, ebi, aebuf, RE_, 576, 128, EH_, 0);  // edge attn out (leaky(ne))

    // residual + LN1
    add_ln<<<(RN_ * 32 + 255) / 256, 256, 0, stream>>>(xbuf, nbuf, l1ng, l1nb, nbuf, RN_, NH_, 1);
    add_ln<<<(RE_ * 32 + 255) / 256, 256, 0, stream>>>(aebuf, ebuf, l1eg, l1eb, aebuf, RE_, EH_, 0);

    // MLPs + residual + LN2
    gemm(nbuf, m1w, m1b, wvb,  RN_, 576, 576, NH_, 1);
    gemm(wvb,  m2w, m2b, xbuf, RN_, 576, 576, NH_, 0);
    add_ln<<<(RN_ * 32 + 255) / 256, 256, 0, stream>>>(xbuf, nbuf, l2ng, l2nb, nbuf, RN_, NH_, 1);
    gemm(aebuf, e1w, e1b, tebuf, RE_, 128, 128, EH_, 1);
    gemm(tebuf, e2w, e2b, ebuf,  RE_, 128, 128, EH_, 0);
    add_ln<<<(RE_ * 32 + 255) / 256, 256, 0, stream>>>(ebuf, aebuf, l2eg, l2eb, ebuf, RE_, EH_, 0);
  }

  // ---- output heads; d_out = [out_n (512x45) | out_e (32768x5)] ----
  float* outp = (float*)d_out;
  gemm(nbuf, t_on1, out_n1_b, xbuf, RN_, 576, 576, NH_, 1);
  gemm(xbuf, t_on2, out_n2_b, outp, RN_, 576, 64, NODE_DIM_, 0);
  gemm(ebuf, t_oe1, out_e1_b, tebuf, RE_, 128, 128, EH_, 1);
  gemm(tebuf, t_oe2, out_e2_b, outp + (size_t)RN_ * NODE_DIM_, RE_, 128, 64, EDGE_DIM_, 0);
}
